// TwoChannelTraceLIF_43843026157720
// MI455X (gfx1250) — matrix operand order repaired and checked by reading, not compiled, not run on hardware
//
#include <hip/hip_runtime.h>
#include <hip/hip_bf16.h>

typedef _Float16     v16h __attribute__((ext_vector_type(16)));
typedef float        v8f  __attribute__((ext_vector_type(8)));
typedef float        f4   __attribute__((ext_vector_type(4)));
typedef unsigned int u4   __attribute__((ext_vector_type(4)));

#define BSZ   16
#define LIN   16384
#define LOUT  16320                      // 16384 - 64 = 51 * 320
#define CCH   64
#define KPAD  96                         // 65 real taps zero-padded to 3 x 32
#define TPB   320                        // output timesteps per conv block
#define NTILE 5                          // 16-wide tiles per wave (4*5*16 = 320)
#define N1    (16LL * 16320LL * 64LL)    // elements per output tensor

// ---------------------------------------------------------------------------
// Prep kernel (1 block): build both signals' zero-padded 64x96 f16 weight
// matrices and fused 64-entry bias vectors into d_ws. Runs once per launch so
// the 1632 conv blocks never redo the branchy f32->f16 conversion.
//   Wp[0      .. 6143]  : x-filterbank weights (row-major [c][96], f16)
//   Wp[6144   .. 12287] : y-filterbank weights
//   bp[0..63] / bp[64..127] : x / y biases (f32)
// ---------------------------------------------------------------------------
__global__ __launch_bounds__(128) void prep_weights_kernel(
    const float* __restrict__ Wx1, const float* __restrict__ bx1,
    const float* __restrict__ Wx2, const float* __restrict__ bx2,
    const float* __restrict__ Wy1, const float* __restrict__ by1,
    const float* __restrict__ Wy2, const float* __restrict__ by2,
    _Float16* __restrict__ Wp, float* __restrict__ bp)
{
    const int tid = threadIdx.x;
    for (int i = tid; i < CCH * KPAD; i += 128) {
        const int c = i / KPAD, j = i % KPAD;
        float wx = 0.0f, wy = 0.0f;
        if (c < 32) {
            if (j >= 32 && j < 65) { wx = Wx1[c * 33 + (j - 32)];
                                     wy = Wy1[c * 33 + (j - 32)]; }
        } else {
            if (j < 65)            { wx = Wx2[(c - 32) * 65 + j];
                                     wy = Wy2[(c - 32) * 65 + j]; }
        }
        Wp[i]              = (_Float16)wx;
        Wp[CCH * KPAD + i] = (_Float16)wy;
    }
    if (tid < 64)       bp[tid] = (tid < 32) ? bx1[tid] : bx2[tid - 32];
    else if (tid < 128) { const int t = tid - 64;
                          bp[tid] = (t < 32) ? by1[t] : by2[t - 32]; }
}

// ---------------------------------------------------------------------------
// Conv kernel: filterbank as GEMM via WMMA.
// D[c,u] = relu( sum_{j=0..64} Wpad[c,j] * x[b, u+j] + bias[c] )
// 128-thread block = 4 waves; block covers 320 timesteps of one batch row.
// Each wave computes five 64(M) x 16(N) tiles, 12 v_wmma_f32_16x16x32_f16
// per tile, from LDS-staged f16 weights (128-bit copies) and signal window.
// ---------------------------------------------------------------------------
__global__ __launch_bounds__(128) void conv_wmma_kernel(
    const float*    __restrict__ x,  const _Float16* __restrict__ Wp,
    const float*    __restrict__ bp, float*          __restrict__ d2out)
{
    __shared__ _Float16 Wl[CCH * KPAD];     // 12 KB padded weights
    __shared__ _Float16 xh[TPB + KPAD];     // 416-sample signal window
    __shared__ float    bl[CCH];

    const int tid = threadIdx.x;
    const int tb  = blockIdx.x % 51;        // time-block within batch row
    const int bb  = blockIdx.x / 51;        // batch row
    const int u0  = tb * TPB;

    // Stage weights with pure 128-bit copies (6 per thread).
    {
        const u4* __restrict__ src = (const u4*)Wp;
        u4*       __restrict__ dst = (u4*)Wl;
        for (int i = tid; i < (CCH * KPAD) / 8; i += 128) dst[i] = src[i];
    }
    // Stage the signal window (320 outputs + 96 taps lookahead, clamped).
    for (int i = tid; i < TPB + KPAD; i += 128) {
        const int gi = u0 + i;
        xh[i] = (_Float16)((gi < LIN) ? x[(size_t)bb * LIN + gi] : 0.0f);
    }
    if (tid < CCH) bl[tid] = bp[tid];
    __syncthreads();

    const int wave = tid >> 5;
    const int lane = tid & 31;
    const int half = lane >> 4;             // 0: lanes 0-15, 1: lanes 16-31
    const int lm   = lane & 15;

    for (int tt = 0; tt < NTILE; ++tt) {
        const int nloc = (tt * 4 + wave) * 16;   // tile base within block

        v8f acc[4];
#pragma unroll
        for (int m = 0; m < 4; ++m) acc[m] = (v8f){0,0,0,0,0,0,0,0};

#pragma unroll
        for (int ks = 0; ks < 3; ++ks) {
            // B fragment (32x16 f16): lane = column n=lm, K = half*16 + i.
            // B[k,n] = xh[nloc + n + ks*32 + k]
            v16h bf;
            const int bbase = nloc + lm + ks * 32 + half * 16;
#pragma unroll
            for (int i = 0; i < 16; ++i) bf[i] = xh[bbase + i];

#pragma unroll
            for (int m = 0; m < 4; ++m) {
                // A fragment (16x32 f16): row M=lm; elems 0..7 -> K=kl+i,
                // elems 8..15 -> K=kh+i (K = half*16 + i, matches B). 16B-aligned runs.
                v16h af;
                const int cw = (m * 16 + lm) * KPAD;
                const int kl = ks * 32 + half * 16;
                const int kh = ks * 32 + half * 16 + 8;
#pragma unroll
                for (int i = 0; i < 8; ++i) {
                    af[i]     = Wl[cw + kl + i];
                    af[8 + i] = Wl[cw + kh + i];
                }
                acc[m] = __builtin_amdgcn_wmma_f32_16x16x32_f16(
                    false, af, false, bf, (short)0, acc[m], false, false);
            }
        }

        // D layout: VGPR r -> M = r + half*8, N = lm. Bias + ReLU, then two
        // 16B vector stores per m-tile (8 consecutive channels).
        const int u = u0 + nloc + lm;
        float* __restrict__ outp = d2out + ((size_t)bb * LOUT + u) * CCH;
#pragma unroll
        for (int m = 0; m < 4; ++m) {
            const int c0 = m * 16 + half * 8;
            f4 vlo, vhi;
#pragma unroll
            for (int r = 0; r < 4; ++r) {
                float a = acc[m][r]     + bl[c0 + r];
                float b = acc[m][r + 4] + bl[c0 + 4 + r];
                vlo[r] = a > 0.0f ? a : 0.0f;
                vhi[r] = b > 0.0f ? b : 0.0f;
            }
            *(f4*)(outp + c0)     = vlo;
            *(f4*)(outp + c0 + 4) = vhi;
        }
    }
}

// ---------------------------------------------------------------------------
// Fused trace + LIF forward. One thread per (signal, batch, channel) sequence
// = 2048 threads (64 wave32s). Forward value of the surrogate spike is exactly
// the hard spike (stop_gradient identity), so no sigmoid/exp needed.
// ---------------------------------------------------------------------------
__global__ __launch_bounds__(256) void trace_lif_kernel(float* __restrict__ out)
{
    const int gid = blockIdx.x * blockDim.x + threadIdx.x;
    if (gid == 0) {                       // warmup scalars (= KMAX-1 = 64)
        out[4 * N1]     = 64.0f;
        out[8 * N1 + 1] = 64.0f;
    }
    if (gid >= 2048) return;

    const int s = gid >> 10;              // signal: 0 = x, 1 = y
    const int r = gid & 1023;
    const int b = r >> 6;
    const int k = r & 63;

    const size_t sbase = (size_t)s * (4 * N1 + 1);
    const float* __restrict__ d2 = out + sbase;        // conv output
    float* __restrict__ zo = out + sbase + 1 * N1;     // trace
    float* __restrict__ so = out + sbase + 2 * N1;     // spikes
    float* __restrict__ eo = out + sbase + 3 * N1;     // running count

    size_t idx = (size_t)b * LOUT * CCH + k;
    float zv = 0.0f, v = 0.0f, a = 0.0f, sn = 0.0f;

    for (int t = 0; t < LOUT; ++t, idx += CCH) {
        __builtin_prefetch(d2 + idx + 64 * 48, 0, 0);  // global_prefetch_b8
        const float d = d2[idx];
        zv = 0.95f * zv + 0.05f * d;                   // trace
        zo[idx] = zv;

        const float th = 1.5f + 1.5f * a;              // adaptive threshold
        v = v * (1.0f - 1.0f / 20.0f) + zv;            // leaky integrate
        const float sh = (v >= th) ? 1.0f : 0.0f;      // hard spike
        sn += sh;
        so[idx] = sh;
        eo[idx] = sn;
        v = (v >= th) ? -0.5f : v;                     // reset
        a = a * (1.0f - 1.0f / 100.0f) + sh;           // adaptation
    }
}

// ---------------------------------------------------------------------------
extern "C" void kernel_launch(void* const* d_in, const int* in_sizes, int n_in,
                              void* d_out, int out_size, void* d_ws, size_t ws_size,
                              hipStream_t stream)
{
    const float* x   = (const float*)d_in[0];
    const float* y   = (const float*)d_in[1];
    const float* Wx1 = (const float*)d_in[2];
    const float* bx1 = (const float*)d_in[3];
    const float* Wx2 = (const float*)d_in[4];
    const float* bx2 = (const float*)d_in[5];
    const float* Wy1 = (const float*)d_in[6];
    const float* by1 = (const float*)d_in[7];
    const float* Wy2 = (const float*)d_in[8];
    const float* by2 = (const float*)d_in[9];

    float* out = (float*)d_out;
    float* d2x = out;                          // offset 0
    float* d2y = out + (4 * N1 + 1);           // after x-tuple + warmup scalar

    _Float16* wsW = (_Float16*)d_ws;           // 2 * 6144 f16 = 24 KB
    float*    wsB = (float*)((char*)d_ws + 2 * CCH * KPAD * sizeof(_Float16));

    prep_weights_kernel<<<1, 128, 0, stream>>>(Wx1, bx1, Wx2, bx2,
                                               Wy1, by1, Wy2, by2, wsW, wsB);

    const int grid = BSZ * 51;                 // 16 batch rows x 51 time blocks
    conv_wmma_kernel<<<grid, 128, 0, stream>>>(x, wsW,              wsB,      d2x);
    conv_wmma_kernel<<<grid, 128, 0, stream>>>(y, wsW + CCH * KPAD, wsB + 64, d2y);

    trace_lif_kernel<<<8, 256, 0, stream>>>(out);
}
